// ScaledDotProductAttentionSafe_31842887533219
// MI455X (gfx1250) — matrix operand order repaired and checked by reading, not compiled, not run on hardware
//
#include <hip/hip_runtime.h>

typedef __attribute__((ext_vector_type(16))) __bf16 v16bf;
typedef __attribute__((ext_vector_type(8)))  __bf16 v8bf;
typedef __attribute__((ext_vector_type(8)))  float  v8f;

#define B_    64
#define LQ_   1024
#define LK_   1024
#define D_    64
// exp(s/8) == exp2(s * (log2e/8)); fold both constants into one multiply
#define SCL2  0.18033688011112042592f
#define WAVES 8
#define ROWE  1032         // 1024 + 8 pad -> 2064B row stride, 16B aligned
#define VTP   40           // V-tile row stride (bf16): 80B, 16B aligned, 20-bank lane stride

// Native bf16 conversions -> v_cvt_pk_bf16_f32 (verified in round-2 asm)
__device__ __forceinline__ __bf16 f2bf(float x) { return (__bf16)x; }
__device__ __forceinline__ float  bf2f(__bf16 h) { return (float)h; }

__device__ __forceinline__ v16bf cvt16(float4 f0, float4 f1, float4 f2, float4 f3) {
  v16bf a;
  a[0]=f2bf(f0.x);  a[1]=f2bf(f0.y);  a[2]=f2bf(f0.z);  a[3]=f2bf(f0.w);
  a[4]=f2bf(f1.x);  a[5]=f2bf(f1.y);  a[6]=f2bf(f1.z);  a[7]=f2bf(f1.w);
  a[8]=f2bf(f2.x);  a[9]=f2bf(f2.y);  a[10]=f2bf(f2.z); a[11]=f2bf(f2.w);
  a[12]=f2bf(f3.x); a[13]=f2bf(f3.y); a[14]=f2bf(f3.z); a[15]=f2bf(f3.w);
  return a;
}

// A operand (16x32 bf16): lane = M row, elem j -> K = 16*hi + j (16 contiguous K values)
__device__ __forceinline__ v16bf pack_a(const float* __restrict__ row, int hi) {
  float4 f0 = *reinterpret_cast<const float4*>(row + hi*16);
  float4 f1 = *reinterpret_cast<const float4*>(row + hi*16 + 4);
  float4 f2 = *reinterpret_cast<const float4*>(row + hi*16 + 8);
  float4 f3 = *reinterpret_cast<const float4*>(row + hi*16 + 12);
  return cvt16(f0, f1, f2, f3);
}

// B operand (32x16 bf16): lane = N col, elem j -> K = 16*hi + j (16 contiguous K values)
__device__ __forceinline__ v16bf pack_b(const float* __restrict__ row, int hi) {
  const float* p = row + 16*hi;
  float4 f0 = *reinterpret_cast<const float4*>(p);
  float4 f1 = *reinterpret_cast<const float4*>(p + 4);
  float4 f2 = *reinterpret_cast<const float4*>(p + 8);
  float4 f3 = *reinterpret_cast<const float4*>(p + 12);
  return cvt16(f0, f1, f2, f3);
}

__device__ __forceinline__ v16bf join8(v8bf lo, v8bf hi8) {
  v16bf a;
#pragma unroll
  for (int j = 0; j < 8; ++j) { a[j] = lo[j]; a[j+8] = hi8[j]; }
  return a;
}

__global__ __launch_bounds__(256)
void attn_fused_kernel(const float* __restrict__ Q, const float* __restrict__ K,
                       const float* __restrict__ V, const unsigned char* __restrict__ Msk,
                       float* __restrict__ Out, float* __restrict__ Attn)
{
  __shared__ __align__(16) __bf16 ldsE[WAVES][16][ROWE];  // unnormalized exp tile per wave
  __shared__ __align__(16) __bf16 ldsVT[D_][VTP];         // transposed V chunk (shared)
  __shared__ float ldsInv[WAVES][16];                     // 1/rowsum per local row

  const int wave = threadIdx.x >> 5;
  const int lane = threadIdx.x & 31;
  const int ln   = lane & 15;
  const int hi   = lane >> 4;

  const int b  = blockIdx.x >> 3;            // batch
  const int qt = blockIdx.x & 7;             // 128-row query tile
  const int m0 = qt*128 + wave*16;           // this wave's 16 query rows

  // ---------------- Phase 1: S = QK^T/8, mask, exp -> LDS (+ row sums) ----------------
  const float* qrow = Q + ((size_t)b*LQ_ + (m0 + ln))*D_;
  const v16bf aq0 = pack_a(qrow,      hi);   // d 0..31
  const v16bf aq1 = pack_a(qrow + 32, hi);   // d 32..63

  float rs[8];
#pragma unroll
  for (int r = 0; r < 8; ++r) rs[r] = 0.0f;

  const unsigned char* mbase = Msk + ((size_t)b*LQ_ + m0)*LK_;

#pragma unroll 1
  for (int nt = 0; nt < LK_/16; ++nt) {
    const float* krow = K + ((size_t)b*LK_ + nt*16 + ln)*D_;
    __builtin_prefetch(krow + 16*D_, 0, 1);  // next K tile -> global_prefetch
    const v16bf bk0 = pack_b(krow,      hi);
    const v16bf bk1 = pack_b(krow + 32, hi);

    v8f c = {0.f,0.f,0.f,0.f,0.f,0.f,0.f,0.f};
    c = __builtin_amdgcn_wmma_f32_16x16x32_bf16(false, aq0, false, bk0, (short)0, c, false, false);
    c = __builtin_amdgcn_wmma_f32_16x16x32_bf16(false, aq1, false, bk1, (short)0, c, false, false);

    const int n = nt*16 + ln;                // C layout: vgpr r -> M = r + 8*hi, N = ln
#pragma unroll
    for (int r = 0; r < 8; ++r) {
      const int ml = r + 8*hi;
      const unsigned char mm = mbase[(size_t)ml*LK_ + n];
      // raw v_exp_f32: args are tiny, no subnormal fixup needed; stays branchless
      const float e = mm ? 0.0f : __builtin_amdgcn_exp2f(c[r] * SCL2);
      rs[r] += e;
      ldsE[wave][ml][n] = f2bf(e);
    }
  }

  // ---------------- Phase 2: row-sum reduce across the 16-lane half, invert ------------
  float inv[8];
#pragma unroll
  for (int r = 0; r < 8; ++r) {
    float s = rs[r];
    s += __shfl_xor(s, 1, 32);
    s += __shfl_xor(s, 2, 32);
    s += __shfl_xor(s, 4, 32);
    s += __shfl_xor(s, 8, 32);
    inv[r] = (s > 0.0f) ? (1.0f / s) : 0.0f;
  }
  if (ln == 0) {
#pragma unroll
    for (int r = 0; r < 8; ++r) ldsInv[wave][hi*8 + r] = inv[r];
  }

  // ---------------- Phase 3: Out = (e @ V) * inv  (V chunk staged via LDS) -------------
  v8f acc[4];
#pragma unroll
  for (int t = 0; t < 4; ++t) acc[t] = (v8f){0.f,0.f,0.f,0.f,0.f,0.f,0.f,0.f};

  const float* vb = V + (size_t)b*LK_*D_;
  const int tlin = threadIdx.x * 8;          // linear float index in 32x64 V chunk
  const int kloc = tlin >> 6;                // staged row (k within chunk)
  const int dv0  = tlin & 63;                // first of 8 contiguous dv columns

#pragma unroll 1
  for (int kc = 0; kc < LK_/32; ++kc) {
    __syncthreads();                         // ldsVT reuse fence (also closes phase 1/2)
    {  // cooperative coalesced stage + transpose of V[kc*32 .. +32)[0..64) -> bf16
      const float* src = vb + ((size_t)(kc*32 + kloc))*D_ + dv0;
      float4 g0 = *reinterpret_cast<const float4*>(src);
      float4 g1 = *reinterpret_cast<const float4*>(src + 4);
      ldsVT[dv0+0][kloc] = f2bf(g0.x);  ldsVT[dv0+1][kloc] = f2bf(g0.y);
      ldsVT[dv0+2][kloc] = f2bf(g0.z);  ldsVT[dv0+3][kloc] = f2bf(g0.w);
      ldsVT[dv0+4][kloc] = f2bf(g1.x);  ldsVT[dv0+5][kloc] = f2bf(g1.y);
      ldsVT[dv0+6][kloc] = f2bf(g1.z);  ldsVT[dv0+7][kloc] = f2bf(g1.w);
    }
    __syncthreads();

    // A operand from ldsE: lane row = ln; elem j -> K = 16*hi + j
    const __bf16* er = &ldsE[wave][ln][0];
    const v8bf alo = *reinterpret_cast<const v8bf*>(er + kc*32 + hi*16);
    const v8bf ahi = *reinterpret_cast<const v8bf*>(er + kc*32 + hi*16 + 8);
    const v16bf ap = join8(alo, ahi);

#pragma unroll
    for (int t = 0; t < 4; ++t) {            // B lane = N = dv col; elem j -> K = 16*hi + j
      const __bf16* vr = &ldsVT[t*16 + ln][16*hi];
      const v8bf b0 = *reinterpret_cast<const v8bf*>(vr);
      const v8bf b1 = *reinterpret_cast<const v8bf*>(vr + 8);
      const v16bf bv = join8(b0, b1);
      acc[t] = __builtin_amdgcn_wmma_f32_16x16x32_bf16(false, ap, false, bv, (short)0, acc[t], false, false);
    }
  }

#pragma unroll
  for (int t = 0; t < 4; ++t) {
#pragma unroll
    for (int r = 0; r < 8; ++r) {            // C layout: M = r + 8*hi, inv[r] matches
      const int m = m0 + r + 8*hi;
      Out[((size_t)b*LQ_ + m)*D_ + t*16 + ln] = acc[t][r] * inv[r];
    }
  }

  // ---------------- Phase 4: coalesced f32 attn = e * inv ------------------------------
  float* arow = Attn + ((size_t)b*LQ_ + m0)*LK_;
#pragma unroll 1
  for (int mi = 0; mi < 16; ++mi) {
    const float invm = ldsInv[wave][mi];
    float* dst = arow + (size_t)mi*LK_;
    const __bf16* src = &ldsE[wave][mi][0];
#pragma unroll
    for (int c0 = lane*8; c0 < LK_; c0 += 256) {
      const v8bf ev = *reinterpret_cast<const v8bf*>(src + c0);
      float4 o0, o1;
      o0.x = bf2f(ev[0])*invm; o0.y = bf2f(ev[1])*invm;
      o0.z = bf2f(ev[2])*invm; o0.w = bf2f(ev[3])*invm;
      o1.x = bf2f(ev[4])*invm; o1.y = bf2f(ev[5])*invm;
      o1.z = bf2f(ev[6])*invm; o1.w = bf2f(ev[7])*invm;
      *reinterpret_cast<float4*>(dst + c0)     = o0;
      *reinterpret_cast<float4*>(dst + c0 + 4) = o1;
    }
  }
}

extern "C" void kernel_launch(void* const* d_in, const int* in_sizes, int n_in,
                              void* d_out, int out_size, void* d_ws, size_t ws_size,
                              hipStream_t stream) {
  (void)in_sizes; (void)n_in; (void)d_ws; (void)ws_size; (void)out_size;
  const float*         q = (const float*)d_in[0];
  const float*         k = (const float*)d_in[1];
  const float*         v = (const float*)d_in[2];
  const unsigned char* m = (const unsigned char*)d_in[3];  // jnp.bool_ = 1 byte/elem
  float* out  = (float*)d_out;
  float* attn = out + (size_t)B_*LQ_*D_;                   // tuple order: (out, attn)
  dim3 grid(B_ * (LQ_/128));   // 512 workgroups
  dim3 block(256);             // 8 wave32 waves
  hipLaunchKernelGGL(attn_fused_kernel, grid, block, 0, stream, q, k, v, m, out, attn);
}
